// LinearPreMix_1039382086403
// MI455X (gfx1250) — matrix operand order repaired and checked by reading, not compiled, not run on hardware
//
#include <hip/hip_runtime.h>

// ---------------------------------------------------------------------------
// Fused QKV projection for MI455X (gfx1250, wave32, WMMA).
//   x     : [B=4, S=4096, E=2048] fp32      -> A [M=16384, K=2048]
//   W_qkv : [3E=6144, E=2048]   fp32        -> B [N=6144,  K=2048]  (A·B^T)
//   out   : q,k,v each [B, H=16, S, DK=128] fp32, concatenated flat.
//
// Preferred path (ws_size >= 92.3 MB):
//   pass 1: fp32 -> bf16 (RNE) into d_ws (x+W bf16 = 92 MB, fully L2-resident)
//   pass 2: double-buffered GEMM, global->LDS via GLOBAL_LOAD_ASYNC_TO_LDS_B128
//           (ASYNCcnt + s_wait_asynccnt), v_wmma_f32_16x16x32_bf16 accum fp32.
// Fallback path: single-pass GEMM with fused fp32->bf16 conversion in staging.
// ---------------------------------------------------------------------------

typedef __attribute__((ext_vector_type(16))) __bf16 v16bf;
typedef __attribute__((ext_vector_type(8)))  float  v8f;
typedef __attribute__((ext_vector_type(4)))  int    v4i;

constexpr int Bq  = 4;
constexpr int Sq  = 4096;
constexpr int Eq  = 2048;
constexpr int Hq  = 16;
constexpr int DKq = Eq / Hq;          // 128
constexpr int Mq  = Bq * Sq;          // 16384
constexpr int Nq  = 3 * Eq;           // 6144
constexpr int Kq  = Eq;               // 2048

constexpr int BT   = 128;             // block tile (M and N)

// bf16 async kernel tiling
constexpr int KT   = 64;              // K elems staged per tile
constexpr int NKT  = Kq / KT;         // 32 K-tiles
constexpr int LDT  = 72;              // LDS pitch in bf16 elems (144 B: 16B-aligned, bank-friendly)

// fused-conversion fallback kernel tiling
constexpr int KTF  = 32;
constexpr int LDTF = 40;              // 80 B pitch

// -------------------- gfx1250 async global->LDS (guarded) -------------------
#if defined(__has_builtin)
#if __has_builtin(__builtin_amdgcn_global_load_async_to_lds_b128)
#define HAVE_ASYNC_LDS 1
#endif
#if __has_builtin(__builtin_amdgcn_s_wait_asynccnt)
#define HAVE_WAIT_ASYNC 1
#endif
#endif

__device__ __forceinline__ void async_copy16(const unsigned short* g, unsigned short* l) {
#ifdef HAVE_ASYNC_LDS
  __builtin_amdgcn_global_load_async_to_lds_b128(
      (__attribute__((address_space(1))) v4i*)g,
      (__attribute__((address_space(3))) v4i*)l,
      0, 0);
#else
  *reinterpret_cast<uint4*>(l) = *reinterpret_cast<const uint4*>(g);
#endif
}

template <int N>
__device__ __forceinline__ void wait_asynccnt() {
#ifdef HAVE_WAIT_ASYNC
  __builtin_amdgcn_s_wait_asynccnt(N);
#endif
}

// ------------------------------- helpers ------------------------------------
__device__ __forceinline__ unsigned int pack2_bf16(float lo, float hi) {
  unsigned int a = __float_as_uint(lo);
  unsigned int b = __float_as_uint(hi);
  a = (a + 0x7FFFu + ((a >> 16) & 1u)) >> 16;   // round-to-nearest-even
  b = (b + 0x7FFFu + ((b >> 16) & 1u)) >> 16;
  return a | (b << 16);
}

union Frag {
  uint4 q[2];
  v16bf v;
};

__device__ __forceinline__ v8f wmma_bf16(const Frag& a, const Frag& b, v8f c) {
  return __builtin_amdgcn_wmma_f32_16x16x32_bf16(false, a.v, false, b.v,
                                                 (short)0, c, false, false);
}

// One K=32 MMA step for a wave's 32x64 sub-tile from LDS tiles with pitch PITCH.
// A-frag layout (16-bit A 16x32): lane<16: elems0-7=K0-7, elems8-15=K16-23;
// lane>=16: elems0-7=K8-15, elems8-15=K24-31.  B is row-striped [n][k].
template <int PITCH>
__device__ __forceinline__ void mma_step(const unsigned short* __restrict__ As,
                                         const unsigned short* __restrict__ Bs,
                                         int wm, int wn, int lr, int half, int ks,
                                         v8f (&acc)[2][4]) {
  Frag fa0, fa1;
  const uint4* pA0 = reinterpret_cast<const uint4*>(As + (wm + lr) * PITCH + ks + half * 8);
  const uint4* pA1 = reinterpret_cast<const uint4*>(As + (wm + 16 + lr) * PITCH + ks + half * 8);
  fa0.q[0] = pA0[0];  fa0.q[1] = pA0[2];        // second K half is +32 B
  fa1.q[0] = pA1[0];  fa1.q[1] = pA1[2];
#pragma unroll
  for (int ni = 0; ni < 4; ++ni) {
    Frag fb;
    const uint4* pB = reinterpret_cast<const uint4*>(Bs + (wn + ni * 16 + lr) * PITCH + ks + half * 8);
    fb.q[0] = pB[0];
    fb.q[1] = pB[2];
    acc[0][ni] = wmma_bf16(fa0, fb, acc[0][ni]);
    acc[1][ni] = wmma_bf16(fa1, fb, acc[1][ni]);
  }
}

// Scatter a wave's accumulators into the [3][B,H,S,DK] output layout.
// C/D layout: VGPR r, lanes 0-15 -> M=r; lanes 16-31 -> M=r+8; N = lane%16.
__device__ __forceinline__ void store_acc(float* __restrict__ out,
                                          int mBase, int nBase, int wm, int wn,
                                          int lr, int half, v8f (&acc)[2][4]) {
#pragma unroll
  for (int mi = 0; mi < 2; ++mi) {
    const int m0 = mBase + wm + mi * 16 + half * 8;
#pragma unroll
    for (int ni = 0; ni < 4; ++ni) {
      const int nG    = nBase + wn + ni * 16 + lr;
      const int which = nG >> 11;            // /E : 0=q, 1=k, 2=v
      const int e     = nG & (Eq - 1);
      const int h     = e >> 7;              // /DK
      const int dk    = e & (DKq - 1);
      const size_t nPart = (size_t)which * ((size_t)Bq * Sq * Eq)
                         + (size_t)h * ((size_t)Sq * DKq)
                         + (size_t)dk;
#pragma unroll
      for (int r = 0; r < 8; ++r) {
        const int mG = m0 + r;
        const int b  = mG >> 12;             // /S
        const int s  = mG & (Sq - 1);
        out[nPart + (size_t)b * ((size_t)Hq * Sq * DKq) + (size_t)s * DKq] =
            acc[mi][ni][r];
      }
    }
  }
}

// ---------------------- pass 1: fp32 -> bf16 conversion ----------------------
__global__ __launch_bounds__(256)
void convert_f32_to_bf16(const float* __restrict__ src,
                         unsigned short* __restrict__ dst, int n8) {
  const int i = blockIdx.x * blockDim.x + threadIdx.x;
  if (i < n8) {
    const float4* p = reinterpret_cast<const float4*>(src + (size_t)i * 8);
    float4 a = p[0], b = p[1];
    uint4 u;
    u.x = pack2_bf16(a.x, a.y);
    u.y = pack2_bf16(a.z, a.w);
    u.z = pack2_bf16(b.x, b.y);
    u.w = pack2_bf16(b.z, b.w);
    reinterpret_cast<uint4*>(dst)[i] = u;
  }
}

// ------------- pass 2: bf16 GEMM, async double-buffered staging --------------
__global__ __launch_bounds__(256)
void qkv_wmma_gemm_bf16(const unsigned short* __restrict__ xb,
                        const unsigned short* __restrict__ wb,
                        float* __restrict__ out) {
  __shared__ unsigned short As[2][BT * LDT];   // 2 x 18432 B
  __shared__ unsigned short Bs[2][BT * LDT];   // 2 x 18432 B

  const int t     = threadIdx.x;               // 0..255
  const int nBase = blockIdx.x * BT;
  const int mBase = blockIdx.y * BT;

  // staging: thread t owns one 64-B half-row (32 bf16) of A and of B per tile
  const int srow = t >> 1;                     // 0..127
  const int scol = (t & 1) * 32;               // 0 or 32 elems

  const int wave = t >> 5;
  const int lane = t & 31;
  const int half = lane >> 4;
  const int lr   = lane & 15;
  const int wm   = (wave & 3) * 32;
  const int wn   = (wave >> 2) * 64;

  v8f acc[2][4];
  const v8f vzero = {0.f, 0.f, 0.f, 0.f, 0.f, 0.f, 0.f, 0.f};
#pragma unroll
  for (int mi = 0; mi < 2; ++mi)
#pragma unroll
    for (int ni = 0; ni < 4; ++ni)
      acc[mi][ni] = vzero;

  const unsigned short* gA = xb + (size_t)(mBase + srow) * Kq + scol;
  const unsigned short* gB = wb + (size_t)(nBase + srow) * Kq + scol;
  unsigned short* lA0 = &As[0][srow * LDT + scol];
  unsigned short* lA1 = &As[1][srow * LDT + scol];
  unsigned short* lB0 = &Bs[0][srow * LDT + scol];
  unsigned short* lB1 = &Bs[1][srow * LDT + scol];

  // prologue: stage tile 0 into buffer 0 (8 async b128 per thread)
#pragma unroll
  for (int c = 0; c < 4; ++c) async_copy16(gA + c * 8, lA0 + c * 8);
#pragma unroll
  for (int c = 0; c < 4; ++c) async_copy16(gB + c * 8, lB0 + c * 8);

  for (int kt = 0; kt < NKT; ++kt) {
    const int cur = kt & 1;
    if (kt + 1 < NKT) {                        // stage next tile into other buffer
      const unsigned short* nA = gA + (size_t)(kt + 1) * KT;
      const unsigned short* nB = gB + (size_t)(kt + 1) * KT;
      unsigned short* dA = cur ? lA0 : lA1;
      unsigned short* dB = cur ? lB0 : lB1;
#pragma unroll
      for (int c = 0; c < 4; ++c) async_copy16(nA + c * 8, dA + c * 8);
#pragma unroll
      for (int c = 0; c < 4; ++c) async_copy16(nB + c * 8, dB + c * 8);
      wait_asynccnt<8>();                      // current tile landed; next in flight
    } else {
      wait_asynccnt<0>();
    }
    __syncthreads();

    mma_step<LDT>(As[cur], Bs[cur], wm, wn, lr, half, 0,  acc);
    mma_step<LDT>(As[cur], Bs[cur], wm, wn, lr, half, 32, acc);

    __syncthreads();                           // all waves done reading `cur`
  }

  store_acc(out, mBase, nBase, wm, wn, lr, half, acc);
}

// ----------- fallback: single-pass GEMM with fused fp32->bf16 staging --------
__device__ __forceinline__ void stage16f(const float* __restrict__ g,
                                         unsigned int* __restrict__ l) {
  const float4* p = reinterpret_cast<const float4*>(g);
  float4 f0 = p[0], f1 = p[1], f2 = p[2], f3 = p[3];
  uint4 u0, u1;
  u0.x = pack2_bf16(f0.x, f0.y);  u0.y = pack2_bf16(f0.z, f0.w);
  u0.z = pack2_bf16(f1.x, f1.y);  u0.w = pack2_bf16(f1.z, f1.w);
  u1.x = pack2_bf16(f2.x, f2.y);  u1.y = pack2_bf16(f2.z, f2.w);
  u1.z = pack2_bf16(f3.x, f3.y);  u1.w = pack2_bf16(f3.z, f3.w);
  reinterpret_cast<uint4*>(l)[0] = u0;
  reinterpret_cast<uint4*>(l)[1] = u1;
}

__global__ __launch_bounds__(256)
void qkv_wmma_gemm_fused(const float* __restrict__ x,
                         const float* __restrict__ w,
                         float* __restrict__ out) {
  __shared__ unsigned short As[BT * LDTF];
  __shared__ unsigned short Bs[BT * LDTF];

  const int t     = threadIdx.x;
  const int nBase = blockIdx.x * BT;
  const int mBase = blockIdx.y * BT;

  const int srow = t >> 1;
  const int scol = (t & 1) << 4;

  const int wave = t >> 5;
  const int lane = t & 31;
  const int half = lane >> 4;
  const int lr   = lane & 15;
  const int wm   = (wave & 3) * 32;
  const int wn   = (wave >> 2) * 64;

  v8f acc[2][4];
  const v8f vzero = {0.f, 0.f, 0.f, 0.f, 0.f, 0.f, 0.f, 0.f};
#pragma unroll
  for (int mi = 0; mi < 2; ++mi)
#pragma unroll
    for (int ni = 0; ni < 4; ++ni)
      acc[mi][ni] = vzero;

  const float* aG = x + (size_t)(mBase + srow) * Kq + scol;
  const float* bG = w + (size_t)(nBase + srow) * Kq + scol;
  unsigned int* aL = reinterpret_cast<unsigned int*>(As + srow * LDTF + scol);
  unsigned int* bL = reinterpret_cast<unsigned int*>(Bs + srow * LDTF + scol);

  for (int kt = 0; kt < Kq / KTF; ++kt) {
    stage16f(aG, aL);
    stage16f(bG, bL);
    aG += KTF;
    bG += KTF;
    if (kt + 1 < Kq / KTF) {
      __builtin_prefetch(aG, 0, 0);            // global_prefetch_b8
      __builtin_prefetch(bG, 0, 0);
    }
    __syncthreads();
    mma_step<LDTF>(As, Bs, wm, wn, lr, half, 0, acc);
    __syncthreads();
  }

  store_acc(out, mBase, nBase, wm, wn, lr, half, acc);
}

// ------------------------------- launcher ------------------------------------
extern "C" void kernel_launch(void* const* d_in, const int* in_sizes, int n_in,
                              void* d_out, int out_size, void* d_ws, size_t ws_size,
                              hipStream_t stream) {
  (void)in_sizes; (void)n_in; (void)out_size;
  const float* x = (const float*)d_in[0];
  const float* w = (const float*)d_in[1];
  float* out = (float*)d_out;

  const dim3 grid(Nq / BT, Mq / BT);           // 48 x 128
  const size_t needWs = ((size_t)Mq * Kq + (size_t)Nq * Kq) * sizeof(unsigned short);

  if (ws_size >= needWs && d_ws != nullptr) {
    unsigned short* xb = (unsigned short*)d_ws;
    unsigned short* wb = xb + (size_t)Mq * Kq;
    const int nx8 = (Mq * Kq) / 8;             // 4,194,304
    const int nw8 = (Nq * Kq) / 8;             // 1,572,864
    convert_f32_to_bf16<<<(nx8 + 255) / 256, 256, 0, stream>>>(x, xb, nx8);
    convert_f32_to_bf16<<<(nw8 + 255) / 256, 256, 0, stream>>>(w, wb, nw8);
    qkv_wmma_gemm_bf16<<<grid, 256, 0, stream>>>(xb, wb, out);
  } else {
    qkv_wmma_gemm_fused<<<grid, 256, 0, stream>>>(x, w, out);
  }
}
